// LinearFusedLigerCE_34849364640323
// MI455X (gfx1250) — matrix operand order repaired and checked by reading, not compiled, not run on hardware
//
#include <hip/hip_runtime.h>
#include <hip/hip_bf16.h>
#include <math.h>

// ---------------------------------------------------------------------------
// Fused Linear + CrossEntropy (Liger-style) for gfx1250 (MI455X), wave32.
//   Pass 0: convert X [N,H] and W [V,H] fp32 -> bf16 copies in workspace
//           (one streaming pass, ~20us at 23.3 TB/s; GEMM is ~1 TFLOP, so
//            preconversion is free and lets the GEMM stage via async copies).
//   Pass 1: logits tile GEMM with v_wmma_f32_16x16x32_bf16, LDS staged by
//           GLOBAL_LOAD_ASYNC_TO_LDS_B128 (ASYNCcnt), double-buffered,
//           fused online max / sum-exp per row -> per-chunk partials.
//   Pass 2: merge partials per row (log-sum-exp), exact fp32 target dot,
//           atomic accumulate sum/count; Pass 3 writes loss scalar.
// ---------------------------------------------------------------------------

typedef __attribute__((ext_vector_type(16))) __bf16 v16bf;
typedef __attribute__((ext_vector_type(8)))  __bf16 v8bf;
typedef __attribute__((ext_vector_type(8)))  float  v8f;

#define N_TOK   8192
#define HDIM    2048
#define VOCAB   32000
#define IGNORE_INDEX (-100)

#define TILE_N  128
#define TILE_V  128
#define KTILE   32
#define LDS_K   (KTILE + 8)             // pad 8 bf16 -> 20-bank row stride, conflict-free
#define SPLITV  25
#define CHUNK_V (VOCAB / SPLITV)        // 1280
#define VT_PER_CHUNK (CHUNK_V / TILE_V) // 10

// Issue one per-lane 16B async global->LDS copy (CDNA5, tracked by ASYNCcnt).
__device__ __forceinline__ void async_copy_b128(unsigned lds_byte_addr,
                                                const __bf16* gptr) {
  unsigned long long ga = (unsigned long long)(uintptr_t)gptr;
  asm volatile("global_load_async_to_lds_b128 %0, %1, off"
               :: "v"(lds_byte_addr), "v"(ga) : "memory");
}
__device__ __forceinline__ void wait_async0() {
  asm volatile("s_wait_asynccnt 0x0" ::: "memory");
}

// ---------------------------------------------------------------------------
// Pass 0: fp32 -> bf16 streaming conversion (8 elements / thread).
// ---------------------------------------------------------------------------
__global__ __launch_bounds__(256)
void cvt_f32_to_bf16(const float* __restrict__ src, __bf16* __restrict__ dst,
                     size_t n) {
  size_t i = ((size_t)blockIdx.x * 256 + threadIdx.x) * 8;
  if (i >= n) return;
  const float4 a = *(const float4*)(src + i);
  const float4 b = *(const float4*)(src + i + 4);
  v8bf h = { (__bf16)a.x, (__bf16)a.y, (__bf16)a.z, (__bf16)a.w,
             (__bf16)b.x, (__bf16)b.y, (__bf16)b.z, (__bf16)b.w };
  *(v8bf*)(dst + i) = h;
}

// ---------------------------------------------------------------------------
// Pass 1: per (V-chunk, row-block) bf16 WMMA GEMM + online softmax partials.
// ---------------------------------------------------------------------------
__global__ __launch_bounds__(256)
void ce_partial_kernel(const __bf16* __restrict__ Xb, const __bf16* __restrict__ Wb,
                       float* __restrict__ pm, float* __restrict__ ps) {
  __shared__ __align__(16) __bf16 a_lds[2][TILE_N * LDS_K];  // 2 x 10 KB
  __shared__ __align__(16) __bf16 b_lds[2][TILE_V * LDS_K];  // 2 x 10 KB
  __shared__ float runM[TILE_N];
  __shared__ float runS[TILE_N];
  __shared__ float redM[TILE_N][2];
  __shared__ float redS[TILE_N][2];
  __shared__ float bcastM[TILE_N];

  const int t       = threadIdx.x;
  const int lane    = t & 31;
  const int wave    = t >> 5;        // 8 waves
  const int waveRow = wave >> 1;     // 0..3 -> 32-row slice
  const int waveCol = wave & 1;      // 0..1 -> 64-col slice
  const int l15     = lane & 15;
  const int lHalf   = lane >> 4;

  const int rowBase = blockIdx.y * TILE_N;
  const int chunk   = blockIdx.x;

  if (t < TILE_N) { runM[t] = -INFINITY; runS[t] = 0.0f; }

  // staging map: thread t -> row t>>1, 32B half (t&1); 2 x b128 per matrix
  const int ldRow  = t >> 1;
  const int ldHalf = t & 1;
  const unsigned aBase0 = (unsigned)(uintptr_t)&a_lds[0][0];
  const unsigned bBase0 = (unsigned)(uintptr_t)&b_lds[0][0];
  const unsigned ldsOff = (unsigned)(ldRow * LDS_K + ldHalf * 16) * 2u; // bytes

  for (int vt = 0; vt < VT_PER_CHUNK; ++vt) {
    const int colBase = chunk * CHUNK_V + vt * TILE_V;

    v8f acc[2][4];
#pragma unroll
    for (int i = 0; i < 2; ++i)
#pragma unroll
      for (int j = 0; j < 4; ++j)
        acc[i][j] = (v8f)(0.0f);

    // prologue: stage K-tile 0 into buffer 0
    {
      const __bf16* ga = Xb + (size_t)(rowBase + ldRow) * HDIM + ldHalf * 16;
      const __bf16* gb = Wb + (size_t)(colBase + ldRow) * HDIM + ldHalf * 16;
      const unsigned abuf = aBase0 + ldsOff;
      const unsigned bbuf = bBase0 + ldsOff;
      async_copy_b128(abuf,      ga);
      async_copy_b128(abuf + 16, ga + 8);
      async_copy_b128(bbuf,      gb);
      async_copy_b128(bbuf + 16, gb + 8);
    }

    for (int k0 = 0; k0 < HDIM; k0 += KTILE) {
      const int buf = (k0 / KTILE) & 1;
      wait_async0();     // this wave's copies into buf have landed in LDS
      __syncthreads();   // all waves' copies visible; buf^1 free for reuse

      // issue async staging for the next K-tile into the other buffer
      if (k0 + KTILE < HDIM) {
        const int kn = k0 + KTILE;
        const __bf16* ga = Xb + (size_t)(rowBase + ldRow) * HDIM + kn + ldHalf * 16;
        const __bf16* gb = Wb + (size_t)(colBase + ldRow) * HDIM + kn + ldHalf * 16;
        const unsigned abuf = aBase0 + (buf ^ 1) * (unsigned)(TILE_N * LDS_K * 2) + ldsOff;
        const unsigned bbuf = bBase0 + (buf ^ 1) * (unsigned)(TILE_V * LDS_K * 2) + ldsOff;
        async_copy_b128(abuf,      ga);
        async_copy_b128(abuf + 16, ga + 8);
        async_copy_b128(bbuf,      gb);
        async_copy_b128(bbuf + 16, gb + 8);
      }

      // ---- compute on `buf` (K = 32): 8 WMMAs / wave ----
      v16bf aF[2];
      v16bf bF[4];
#pragma unroll
      for (int i = 0; i < 2; ++i) {  // A frag: lane half h holds K h*8..+7 and 16+h*8..+7
        const int row = waveRow * 32 + i * 16 + l15;
        const __bf16* p = &a_lds[buf][row * LDS_K + lHalf * 8];
        v8bf lo = *(const v8bf*)p;
        v8bf hi = *(const v8bf*)(p + 16);
        aF[i] = __builtin_shufflevector(lo, hi, 0,1,2,3,4,5,6,7,8,9,10,11,12,13,14,15);
      }
#pragma unroll
      for (int j = 0; j < 4; ++j) {  // B frag: lane = column, lane half h holds K h*8..+7 and 16+h*8..+7
        const int col = waveCol * 64 + j * 16 + l15;
        const __bf16* p = &b_lds[buf][col * LDS_K + lHalf * 8];
        v8bf lo = *(const v8bf*)p;
        v8bf hi = *(const v8bf*)(p + 16);
        bF[j] = __builtin_shufflevector(lo, hi, 0,1,2,3,4,5,6,7,8,9,10,11,12,13,14,15);
      }
#pragma unroll
      for (int i = 0; i < 2; ++i)
#pragma unroll
        for (int j = 0; j < 4; ++j)
          acc[i][j] = __builtin_amdgcn_wmma_f32_16x16x32_bf16(
              false, aF[i], false, bF[j], (short)0, acc[i][j], false, false);
    }

    // ---- online softmax update over this 128-col tile ----
    // pass 1: per-row max (C layout: VGPR r, lanes0-15 -> M=r, lanes16-31 -> M=8+r)
#pragma unroll
    for (int i = 0; i < 2; ++i) {
#pragma unroll
      for (int r = 0; r < 8; ++r) {
        float v = acc[i][0][r];
        v = fmaxf(v, acc[i][1][r]);
        v = fmaxf(v, acc[i][2][r]);
        v = fmaxf(v, acc[i][3][r]);
#pragma unroll
        for (int m = 1; m < 16; m <<= 1) v = fmaxf(v, __shfl_xor(v, m, 32));
        if (l15 == 0) {
          const int rowLocal = waveRow * 32 + i * 16 + lHalf * 8 + r;
          redM[rowLocal][waveCol] = v;
        }
      }
    }
    __syncthreads();
    if (t < TILE_N) {
      const float tm   = fmaxf(redM[t][0], redM[t][1]);
      const float newM = fmaxf(runM[t], tm);
      runS[t] *= __expf(runM[t] - newM);  // first tile: exp(-inf)=0
      runM[t]   = newM;
      bcastM[t] = newM;
    }
    __syncthreads();
    // pass 2: per-row sum of exp(l - M)
#pragma unroll
    for (int i = 0; i < 2; ++i) {
#pragma unroll
      for (int r = 0; r < 8; ++r) {
        const int rowLocal = waveRow * 32 + i * 16 + lHalf * 8 + r;
        const float Mr = bcastM[rowLocal];
        float s = __expf(acc[i][0][r] - Mr) + __expf(acc[i][1][r] - Mr) +
                  __expf(acc[i][2][r] - Mr) + __expf(acc[i][3][r] - Mr);
#pragma unroll
        for (int m = 1; m < 16; m <<= 1) s += __shfl_xor(s, m, 32);
        if (l15 == 0) redS[rowLocal][waveCol] = s;
      }
    }
    __syncthreads();
    if (t < TILE_N) runS[t] += redS[t][0] + redS[t][1];
    // a/b tiles only touched again by async issue after next loop's barrier
  }

  __syncthreads();
  if (t < TILE_N) {
    const size_t idx = (size_t)chunk * N_TOK + rowBase + t;
    pm[idx] = runM[t];
    ps[idx] = runS[t];
  }
}

// ---------------------------------------------------------------------------
// Pass 2: per-row merge of SPLITV partials + exact fp32 target logit + NLL.
// One wave per row.
// ---------------------------------------------------------------------------
__global__ __launch_bounds__(256)
void ce_finalize_rows(const float* __restrict__ X, const float* __restrict__ W,
                      const int* __restrict__ Y,
                      const float* __restrict__ pm, const float* __restrict__ ps,
                      float* __restrict__ accum) {
  const int lane = threadIdx.x & 31;
  const int wave = threadIdx.x >> 5;
  const int row  = blockIdx.x * 8 + wave;
  if (row >= N_TOK) return;

  const int  yraw  = Y[row];
  const bool valid = (yraw != IGNORE_INDEX);
  const int  yv    = valid ? yraw : 0;

  // exact fp32 dot: x[row] . W[yv]
  const float4* xp = (const float4*)(X + (size_t)row * HDIM);
  const float4* wp = (const float4*)(W + (size_t)yv  * HDIM);
  float dot = 0.0f;
  for (int q = lane; q < HDIM / 4; q += 32) {
    const float4 a = xp[q];
    const float4 b = wp[q];
    dot = fmaf(a.x, b.x, dot);
    dot = fmaf(a.y, b.y, dot);
    dot = fmaf(a.z, b.z, dot);
    dot = fmaf(a.w, b.w, dot);
  }
#pragma unroll
  for (int m = 1; m < 32; m <<= 1) dot += __shfl_xor(dot, m, 32);

  if (lane == 0) {
    float M = -INFINITY, S = 0.0f;
    for (int c = 0; c < SPLITV; ++c) {
      const float m  = pm[(size_t)c * N_TOK + row];
      const float s  = ps[(size_t)c * N_TOK + row];
      const float nM = fmaxf(M, m);
      S = S * __expf(M - nM) + s * __expf(m - nM);
      M = nM;
    }
    const float nll = M + logf(S) - dot;  // -log_softmax at target
    atomicAdd(&accum[0], valid ? nll : 0.0f);
    atomicAdd(&accum[1], valid ? 1.0f : 0.0f);
  }
}

__global__ void ce_init(float* __restrict__ accum) {
  if (threadIdx.x == 0) { accum[0] = 0.0f; accum[1] = 0.0f; }
}

__global__ void ce_write_loss(const float* __restrict__ accum, float* __restrict__ out) {
  if (threadIdx.x == 0) out[0] = accum[0] / fmaxf(accum[1], 1.0f);
}

// ---------------------------------------------------------------------------
extern "C" void kernel_launch(void* const* d_in, const int* in_sizes, int n_in,
                              void* d_out, int out_size, void* d_ws, size_t ws_size,
                              hipStream_t stream) {
  const float* X = (const float*)d_in[0];  // [N, H] fp32
  const float* W = (const float*)d_in[1];  // [V, H] fp32
  const int*   Y = (const int*)d_in[2];    // [N] labels
  float* out = (float*)d_out;              // scalar loss

  // workspace layout (16B-aligned sections):
  //   pm [SPLITV][N] f32 | ps [SPLITV][N] f32 | accum[2]+pad | Xb bf16 | Wb bf16
  float* ws    = (float*)d_ws;
  float* pm    = ws;
  float* ps    = pm + (size_t)SPLITV * N_TOK;
  float* accum = ps + (size_t)SPLITV * N_TOK;
  __bf16* Xb = (__bf16*)(accum + 4);
  __bf16* Wb = Xb + (size_t)N_TOK * HDIM;

  ce_init<<<1, 32, 0, stream>>>(accum);

  // Pass 0: fp32 -> bf16 copies (streaming, bandwidth-trivial)
  {
    const size_t nx = (size_t)N_TOK * HDIM;   // 16.8M
    const size_t nw = (size_t)VOCAB * HDIM;   // 65.5M
    cvt_f32_to_bf16<<<(unsigned)((nx / 8 + 255) / 256), 256, 0, stream>>>(X, Xb, nx);
    cvt_f32_to_bf16<<<(unsigned)((nw / 8 + 255) / 256), 256, 0, stream>>>(W, Wb, nw);
  }

  // Pass 1: WMMA GEMM + online softmax partials
  dim3 grid1(SPLITV, N_TOK / TILE_N);   // 25 x 64 workgroups
  ce_partial_kernel<<<grid1, 256, 0, stream>>>(Xb, Wb, pm, ps);

  // Pass 2/3: merge + loss
  ce_finalize_rows<<<N_TOK / 8, 256, 0, stream>>>(X, W, Y, pm, ps, accum);
  ce_write_loss<<<1, 32, 0, stream>>>(accum, out);
}
